// Celebellum_86285892977080
// MI455X (gfx1250) — compile-verified, matrix operand order checked by reading, not run on hardware
//
#include <hip/hip_runtime.h>

typedef __attribute__((ext_vector_type(16))) __bf16 v16bf;
typedef __attribute__((ext_vector_type(8)))  __bf16 v8bf;
typedef __attribute__((ext_vector_type(4)))  __bf16 v4bf;
typedef __attribute__((ext_vector_type(8)))  float  v8f;
typedef __attribute__((ext_vector_type(4)))  float  v4f;

#define T_STEPS 16
#define BATCH   16384
#define DIN     512
#define HIDN    512
#define OUTN    7
#define MT      32      // batch rows per workgroup
#define BT      256     // threads per block (8 wave32) -> 2 waves/SIMD, 512-VGPR budget
#define LSTR    520     // LDS row stride (bf16 elems), 16B-aligned rows, skews banks

__device__ __forceinline__ __bf16 f2bf(float f) {
    unsigned int u = __builtin_bit_cast(unsigned int, f);
    u += 0x7FFFu + ((u >> 16) & 1u);            // round-to-nearest-even
    return __builtin_bit_cast(__bf16, (unsigned short)(u >> 16));
}

// ---- weight repack: fp32 [K][N] row-major -> bf16 WMMA-B fragment order ----
// Tiles of 32(K)x16(N); per tile 512 elems laid out lane-major:
//   lane = nn + 16*((kk>>3)&1),  j = (kk&7) + 8*((kk>>4)&1)
__global__ void repack_w(const float* __restrict__ W, __bf16* __restrict__ dst, int K) {
    int idx = blockIdx.x * blockDim.x + threadIdx.x;
    if (idx >= K * 512) return;
    int k = idx >> 9, n = idx & 511;
    int kt = k >> 5, kk = k & 31, nt = n >> 4, nn = n & 15;
    int lane = nn + 16 * ((kk >> 3) & 1);
    int j    = (kk & 7) + 8 * ((kk >> 4) & 1);
    dst[(size_t)(kt * 32 + nt) * 512 + lane * 16 + j] = f2bf(W[idx]);
}

// W_d2: [512][7] -> padded 32x16 K-tiles (cols 7..15 zero)
__global__ void repack_wd2(const float* __restrict__ W, __bf16* __restrict__ dst) {
    int idx = blockIdx.x * blockDim.x + threadIdx.x;
    if (idx >= 512 * 16) return;
    int k = idx >> 4, n = idx & 15;
    int kt = k >> 5, kk = k & 31;
    int lane = n + 16 * ((kk >> 3) & 1);
    int j    = (kk & 7) + 8 * ((kk >> 4) & 1);
    float v = (n < OUTN) ? W[k * OUTN + n] : 0.0f;
    dst[(size_t)kt * 512 + lane * 16 + j] = f2bf(v);
}

// A fragment (16x32 bf16) from LDS: lane 0-15 = rows, half selects K chunk
#define LOAD_A(BUF, ROW, K0)                                                     \
    ({ v8bf lo_ = *(const v8bf*)&BUF[(ROW) * LSTR + (K0) + hl * 8];              \
       v8bf hi_ = *(const v8bf*)&BUF[(ROW) * LSTR + (K0) + 16 + hl * 8];         \
       __builtin_shufflevector(lo_, hi_, 0,1,2,3,4,5,6,7,8,9,10,11,12,13,14,15); })

// B fragment: contiguous 32B per lane from repacked weights (L2 resident)
#define LOAD_B(PTR, TILE) (*(const v16bf*)((PTR) + (size_t)(TILE) * 512 + lane * 16))

#define WMMA_BF16(A, B, C) \
    __builtin_amdgcn_wmma_f32_16x16x32_bf16(false, (A), false, (B), (short)0, (C), false, false)

__global__ __launch_bounds__(BT) void snn_persistent(
    const float* __restrict__ x,
    const float* __restrict__ b_gc, const float* __restrict__ b_pc,
    const float* __restrict__ b_d1, const float* __restrict__ b_d2,
    const __bf16* __restrict__ rWgc, const __bf16* __restrict__ rWpc,
    const __bf16* __restrict__ rWd1, const __bf16* __restrict__ rWd2,
    float* __restrict__ out)
{
    __shared__ __bf16 lds_x[MT * LSTR];   // x_t slab (bf16)
    __shared__ __bf16 lds_s[MT * LSTR];   // spike slab: gc -> pc -> s3

    const int tid  = threadIdx.x;
    const int wv   = tid >> 5;            // wave id 0..7
    const int lane = tid & 31;
    const int hl   = lane >> 4;           // lane half
    const int ln   = lane & 15;
    const int rowbase = blockIdx.x * MT;
    const int n0 = wv * 64;               // this wave's 64-col slice of HID

    const __bf16 BF1 = __builtin_bit_cast(__bf16, (unsigned short)0x3F80);
    const __bf16 BF0 = __builtin_bit_cast(__bf16, (unsigned short)0x0000);

    // membranes resident in registers, WMMA C-layout: [row-tile][col-tile]
    v8f m1[2][4], m2[2][4], m3[2][4];
    #pragma unroll
    for (int rt = 0; rt < 2; ++rt)
        #pragma unroll
        for (int nt = 0; nt < 4; ++nt) { v8f z = {}; m1[rt][nt] = z; m2[rt][nt] = z; m3[rt][nt] = z; }

    float bg[4], bp[4], bd[4];
    #pragma unroll
    for (int nt = 0; nt < 4; ++nt) {
        bg[nt] = b_gc[n0 + nt * 16 + ln];
        bp[nt] = b_pc[n0 + nt * 16 + ln];
        bd[nt] = b_d1[n0 + nt * 16 + ln];
    }
    v8f oacc = {};   // running sum of s3 @ W_d2 over all t (waves 0,1)

    #pragma unroll 1
    for (int t = 0; t < T_STEPS; ++t) {
        // ---- stage x_t slab: fp32 global -> bf16 LDS ----
        const float* xs = x + ((size_t)t * BATCH + rowbase) * DIN;
        for (int i = tid * 4; i < MT * DIN; i += BT * 4) {
            int r = i >> 9, c = i & 511;
            v4f vv = *(const v4f*)(xs + ((size_t)r << 9) + c);
            v4bf bb; bb.x = f2bf(vv.x); bb.y = f2bf(vv.y); bb.z = f2bf(vv.z); bb.w = f2bf(vv.w);
            *(v4bf*)&lds_x[r * LSTR + c] = bb;
        }
        __syncthreads();

        // ---- layer 1: gc = LIF(m1, x @ W_gc + b_gc); spikes -> lds_s ----
        {
            v8f acc[2][4] = {};
            for (int kt = 0; kt < 16; ++kt) {
                int k0 = kt * 32;
                v16bf a0 = LOAD_A(lds_x, ln, k0);
                v16bf a1 = LOAD_A(lds_x, 16 + ln, k0);
                if (kt + 1 < 16)
                    __builtin_prefetch(rWgc + (size_t)((kt + 1) * 32 + (n0 >> 4)) * 512 + lane * 16, 0, 1);
                #pragma unroll
                for (int nt = 0; nt < 4; ++nt) {
                    v16bf b = LOAD_B(rWgc, kt * 32 + (n0 >> 4) + nt);
                    acc[0][nt] = WMMA_BF16(a0, b, acc[0][nt]);
                    acc[1][nt] = WMMA_BF16(a1, b, acc[1][nt]);
                }
            }
            #pragma unroll
            for (int rt = 0; rt < 2; ++rt)
                #pragma unroll
                for (int nt = 0; nt < 4; ++nt)
                    #pragma unroll
                    for (int v = 0; v < 8; ++v) {
                        float I  = acc[rt][nt][v] + bg[nt];
                        float m  = m1[rt][nt][v];
                        float vm = m + (I - m) * 0.5f;
                        bool  s  = vm > 1.0f;
                        m1[rt][nt][v] = s ? 0.0f : vm;
                        lds_s[(rt * 16 + hl * 8 + v) * LSTR + (n0 + nt * 16 + ln)] = s ? BF1 : BF0;
                    }
        }
        __syncthreads();

        // ---- layer 2: pc = LIF(m2, gc @ W_pc + b_pc) ----
        {
            v8f acc[2][4] = {};
            for (int kt = 0; kt < 16; ++kt) {
                int k0 = kt * 32;
                v16bf a0 = LOAD_A(lds_s, ln, k0);
                v16bf a1 = LOAD_A(lds_s, 16 + ln, k0);
                if (kt + 1 < 16)
                    __builtin_prefetch(rWpc + (size_t)((kt + 1) * 32 + (n0 >> 4)) * 512 + lane * 16, 0, 1);
                #pragma unroll
                for (int nt = 0; nt < 4; ++nt) {
                    v16bf b = LOAD_B(rWpc, kt * 32 + (n0 >> 4) + nt);
                    acc[0][nt] = WMMA_BF16(a0, b, acc[0][nt]);
                    acc[1][nt] = WMMA_BF16(a1, b, acc[1][nt]);
                }
            }
            // LIF; keep spike (0/1) in acc, delay LDS overwrite until all reads done
            #pragma unroll
            for (int rt = 0; rt < 2; ++rt)
                #pragma unroll
                for (int nt = 0; nt < 4; ++nt)
                    #pragma unroll
                    for (int v = 0; v < 8; ++v) {
                        float I  = acc[rt][nt][v] + bp[nt];
                        float m  = m2[rt][nt][v];
                        float vm = m + (I - m) * 0.5f;
                        bool  s  = vm > 1.0f;
                        m2[rt][nt][v] = s ? 0.0f : vm;
                        acc[rt][nt][v] = s ? 1.0f : 0.0f;
                    }
            __syncthreads();
            #pragma unroll
            for (int rt = 0; rt < 2; ++rt)
                #pragma unroll
                for (int nt = 0; nt < 4; ++nt)
                    #pragma unroll
                    for (int v = 0; v < 8; ++v)
                        lds_s[(rt * 16 + hl * 8 + v) * LSTR + (n0 + nt * 16 + ln)] =
                            (acc[rt][nt][v] != 0.0f) ? BF1 : BF0;
        }
        __syncthreads();

        // ---- layer 3: s3 = LIF(m3, [x_t, pc] @ W_d1 + b_d1), K = 1024 ----
        {
            v8f acc[2][4] = {};
            for (int kt = 0; kt < 32; ++kt) {
                v16bf a0, a1;
                if (kt < 16) { a0 = LOAD_A(lds_x, ln, kt * 32);        a1 = LOAD_A(lds_x, 16 + ln, kt * 32); }
                else         { a0 = LOAD_A(lds_s, ln, (kt - 16) * 32); a1 = LOAD_A(lds_s, 16 + ln, (kt - 16) * 32); }
                if (kt + 1 < 32)
                    __builtin_prefetch(rWd1 + (size_t)((kt + 1) * 32 + (n0 >> 4)) * 512 + lane * 16, 0, 1);
                #pragma unroll
                for (int nt = 0; nt < 4; ++nt) {
                    v16bf b = LOAD_B(rWd1, kt * 32 + (n0 >> 4) + nt);
                    acc[0][nt] = WMMA_BF16(a0, b, acc[0][nt]);
                    acc[1][nt] = WMMA_BF16(a1, b, acc[1][nt]);
                }
            }
            #pragma unroll
            for (int rt = 0; rt < 2; ++rt)
                #pragma unroll
                for (int nt = 0; nt < 4; ++nt)
                    #pragma unroll
                    for (int v = 0; v < 8; ++v) {
                        float I  = acc[rt][nt][v] + bd[nt];
                        float m  = m3[rt][nt][v];
                        float vm = m + (I - m) * 0.5f;
                        bool  s  = vm > 1.0f;
                        m3[rt][nt][v] = s ? 0.0f : vm;
                        acc[rt][nt][v] = s ? 1.0f : 0.0f;
                    }
            __syncthreads();
            #pragma unroll
            for (int rt = 0; rt < 2; ++rt)
                #pragma unroll
                for (int nt = 0; nt < 4; ++nt)
                    #pragma unroll
                    for (int v = 0; v < 8; ++v)
                        lds_s[(rt * 16 + hl * 8 + v) * LSTR + (n0 + nt * 16 + ln)] =
                            (acc[rt][nt][v] != 0.0f) ? BF1 : BF0;
        }
        __syncthreads();

        // ---- layer 4: out_acc += s3 @ W_d2 (N padded 7->16); waves 0,1 only ----
        if (wv < 2) {
            #pragma unroll
            for (int kt = 0; kt < 16; ++kt) {
                v16bf a = LOAD_A(lds_s, wv * 16 + ln, kt * 32);
                v16bf b = LOAD_B(rWd2, kt);
                oacc = WMMA_BF16(a, b, oacc);
            }
        }
        __syncthreads();   // protect lds_x/lds_s before next timestep's staging
    }

    // ---- epilogue: out = oacc/T + b_d2 ----
    if (wv < 2 && ln < OUTN) {
        #pragma unroll
        for (int v = 0; v < 8; ++v) {
            int row = rowbase + wv * 16 + hl * 8 + v;
            out[(size_t)row * OUTN + ln] = oacc[v] * (1.0f / T_STEPS) + b_d2[ln];
        }
    }
}

extern "C" void kernel_launch(void* const* d_in, const int* in_sizes, int n_in,
                              void* d_out, int out_size, void* d_ws, size_t ws_size,
                              hipStream_t stream) {
    const float* x    = (const float*)d_in[0];
    const float* Wgc  = (const float*)d_in[1];
    const float* bgc  = (const float*)d_in[2];
    const float* Wpc  = (const float*)d_in[3];
    const float* bpc  = (const float*)d_in[4];
    const float* Wd1  = (const float*)d_in[5];
    const float* bd1  = (const float*)d_in[6];
    const float* Wd2  = (const float*)d_in[7];
    const float* bd2  = (const float*)d_in[8];
    float* out = (float*)d_out;

    // workspace carve: repacked bf16 weights (~2.05 MB), L2-resident thereafter
    __bf16* rWgc = (__bf16*)d_ws;
    __bf16* rWpc = rWgc + (size_t)512 * 512;
    __bf16* rWd1 = rWpc + (size_t)512 * 512;
    __bf16* rWd2 = rWd1 + (size_t)1024 * 512;

    repack_w <<<(512 * 512 + 255) / 256, 256, 0, stream>>>(Wgc, rWgc, 512);
    repack_w <<<(512 * 512 + 255) / 256, 256, 0, stream>>>(Wpc, rWpc, 512);
    repack_w <<<(1024 * 512 + 255) / 256, 256, 0, stream>>>(Wd1, rWd1, 1024);
    repack_wd2<<<(512 * 16 + 255) / 256, 256, 0, stream>>>(Wd2, rWd2);

    snn_persistent<<<BATCH / MT, BT, 0, stream>>>(
        x, bgc, bpc, bd1, bd2, rWgc, rWpc, rWd1, rWd2, out);
}
